// Classifier_custom_37280316129862
// MI455X (gfx1250) — compile-verified, matrix operand order checked by reading, not run on hardware
//
#include <hip/hip_runtime.h>

// ---------------------------------------------------------------------------
// CDNA5 (gfx1250) 4-block CNN classifier.
//  * all convs = implicit GEMM on v_wmma_f32_16x16x32_bf16 (bf16 A/B, f32 acc)
//  * activations kept in bf16 (23.3 TB/s HBM -> half traffic vs fp32)
//  * ALL matrix tiles (weight slabs + 1x1 B tiles) staged into LDS by the
//    Tensor Data Mover (tensor_load_to_lds), synchronized with
//    s_wait_tensorcnt + workgroup barriers
// ---------------------------------------------------------------------------

typedef __attribute__((ext_vector_type(16))) __bf16 v16bf;
typedef __attribute__((ext_vector_type(8)))  float  v8f;
typedef unsigned u32;
typedef __attribute__((ext_vector_type(4))) u32 u32x4;
typedef __attribute__((ext_vector_type(8))) u32 u32x8;

__device__ __forceinline__ unsigned short f32_to_bf16(float f) {
  unsigned u = __float_as_uint(f);
  u += 0x7FFFu + ((u >> 16) & 1u);        // round-to-nearest-even
  return (unsigned short)(u >> 16);
}
__device__ __forceinline__ float bf16_to_f32(unsigned short h) {
  return __uint_as_float(((unsigned)h) << 16);
}

// Issue one TDM 2D-tile load: rows x cols tile of a [rows_total][stride]
// u16 tensor whose remaining extent from the tile origin is (dim1 x dim0).
// OOB reads return zero. lds/global addresses in bytes.
__device__ __forceinline__ void tdm_load_2d(unsigned lds_addr,
                                            unsigned long long gaddr,
                                            int dim0_rem, int dim1_rem,
                                            int tile0, int tile1, int stride) {
  u32x4 g0;
  u32x8 g1;
  g0[0] = 1u;                                           // count=1, user mode
  g0[1] = lds_addr;                                     // lds_addr (bytes)
  g0[2] = (u32)gaddr;                                   // global_addr[31:0]
  g0[3] = (u32)((gaddr >> 32) & 0x1FFFFFFu) | (2u << 30);  // [56:32] | type=2
  g1[0] = 0x10000u;                                     // data_size = 2 bytes
  g1[1] = (u32)dim0_rem << 16;                          // tensor_dim0
  g1[2] = (u32)dim1_rem << 16;                          // tensor_dim1
  g1[3] = (u32)tile0 << 16;                             // tile_dim0
  g1[4] = (u32)tile1;                                   // tile_dim1
  g1[5] = (u32)stride;                                  // tensor_dim0_stride
  g1[6] = 0u;
  g1[7] = 0u;
  asm volatile("tensor_load_to_lds %0, %1" :: "s"(g0), "s"(g1) : "memory");
}

// fp32 -> bf16 tensor conversion (grid-stride)
__global__ void k_cast_bf16(const float* __restrict__ x,
                            unsigned short* __restrict__ y, long n) {
  for (long i = (long)blockIdx.x * blockDim.x + threadIdx.x; i < n;
       i += (long)gridDim.x * blockDim.x)
    y[i] = f32_to_bf16(x[i]);
}

// ---------------------------------------------------------------------------
// Generic kxk implicit-GEMM convolution (spatial convs).
// M = Cout, N = batch*Hout*Wout, K = Cin*k*k. 8 waves/block share a 16-row
// Cout weight slab (TDM-loaded into LDS); each wave owns a 16-position tile
// whose B fragment is gathered per-lane (im2col addressing).
// Fragment layout (ISA 7.12.2, 16-bit, K=32): element e -> K-offset
//   kq = e + 8*(lane>>4) + (e>=8 ? 8 : 0), row/col = lane & 15.
// C/D: VGPR r, lane -> M = r + 8*(lane>>4), N = lane & 15.
// ---------------------------------------------------------------------------
__global__ void k_conv_wmma(const unsigned short* __restrict__ in,
                            const unsigned short* __restrict__ w,
                            const float* __restrict__ bias,
                            unsigned short* __restrict__ out,
                            int Nb, int Cin, int Hin, int Win,
                            int Cout, int Hout, int Wout,
                            int ksz, int stride, int pad, int relu) {
  const int lane = threadIdx.x & 31;
  const int wave = threadIdx.x >> 5;
  const int half = lane >> 4;
  const int lm   = lane & 15;
  const int ksq  = ksz * ksz;
  const int K    = Cin * ksq;
  const int Ntot  = Nb * Hout * Wout;
  const int mbase = blockIdx.y * 16;
  const int p0    = (blockIdx.x * 8 + wave) * 16 + lm;

  const bool pvalid = (p0 < Ntot);
  int img = 0, oh = 0, ow = 0;
  if (pvalid) {
    const int hw = Hout * Wout;
    img = p0 / hw;
    const int r = p0 - img * hw;
    oh = r / Wout;
    ow = r - oh * Wout;
  }

  __shared__ unsigned short sW[16 * 32];
  const unsigned lds_w = (unsigned)(unsigned long long)(void*)&sW[0];
  const unsigned long long w_base = (unsigned long long)w;

  v8f acc = {};

  for (int kk = 0; kk < K; kk += 32) {
    // weight slab (16 Cout rows x 32 K) via Tensor Data Mover, wave 0 only
    if (wave == 0)
      tdm_load_2d(lds_w, w_base + 2ull * ((unsigned long long)mbase * K + kk),
                  K - kk, Cout - mbase, 32, 16, K);
    if (kk + 32 < K)
      __builtin_prefetch(w + mbase * K + kk + 32, 0, 1);
    __builtin_amdgcn_s_wait_tensorcnt(0);
    __syncthreads();

    union { v16bf v; unsigned short u[16]; } Af, Bf;
#pragma unroll
    for (int e = 0; e < 16; ++e) {
      const int kq   = e + 8 * half + (e >= 8 ? 8 : 0);
      const int kidx = kk + kq;
      Af.u[e] = sW[lm * 32 + kq];
      unsigned short bv = 0;
      if (kidx < K && pvalid) {
        int ci, kh, kw;
        if (ksz == 1) { ci = kidx; kh = 0; kw = 0; }
        else {
          ci = kidx / ksq;
          const int r = kidx - ci * ksq;
          kh = r / ksz;
          kw = r - kh * ksz;
        }
        const int ih = oh * stride - pad + kh;
        const int iw = ow * stride - pad + kw;
        if (ih >= 0 && ih < Hin && iw >= 0 && iw < Win)
          bv = in[((img * Cin + ci) * Hin + ih) * Win + iw];
      }
      Bf.u[e] = bv;
    }

    acc = __builtin_amdgcn_wmma_f32_16x16x32_bf16(
        false, Af.v, false, Bf.v, (short)0, acc, false, false);
    __syncthreads();
  }

  if (pvalid) {
#pragma unroll
    for (int r = 0; r < 8; ++r) {
      const int co = mbase + r + 8 * half;
      if (co < Cout) {
        float v = acc[r] + bias[co];
        if (relu) v = fmaxf(v, 0.0f);
        out[((img * Cout + co) * Hout + oh) * Wout + ow] = f32_to_bf16(v);
      }
    }
  }
}

// ---------------------------------------------------------------------------
// 1x1 convolution = GEMM  out[co][img,p] = sum_ci w[co][ci] * in[img][ci][p].
// Per-image 32-wide position tiles; each wave's B tile (32 K-rows x 32
// positions, stride = HW) and the block's 32-row weight slab are rectangular
// 2D tiles fetched by the Tensor Data Mover straight into LDS (OOB -> zero).
// 2 Cout tiles x 2 position subtiles => 4 WMMAs per K-chunk per wave.
// ---------------------------------------------------------------------------
__global__ void k_gemm1x1_wmma(const unsigned short* __restrict__ in,
                               const unsigned short* __restrict__ w,
                               const float* __restrict__ bias,
                               unsigned short* __restrict__ out,
                               int Nb, int Cin, int HW, int Cout, int relu) {
  const int lane = threadIdx.x & 31;
  const int wave = threadIdx.x >> 5;
  const int half = lane >> 4;
  const int lm   = lane & 15;

  const int tiles_per_img = (HW + 31) >> 5;
  const int total_tiles   = Nb * tiles_per_img;
  const int tid    = blockIdx.x * 8 + wave;
  const bool wvalid = (tid < total_tiles);
  const int img = wvalid ? tid / tiles_per_img : 0;
  const int x0  = (tid - img * tiles_per_img) * 32;
  const int mbase = blockIdx.y * 32;

  __shared__ unsigned short sW[32 * 32];       // 32 Cout rows x 32 K
  __shared__ unsigned short sB[8][32 * 32];    // per-wave B tile [k][pos]

  const unsigned lds_w = (unsigned)(unsigned long long)(void*)&sW[0];
  const unsigned lds_b = (unsigned)(unsigned long long)(void*)&sB[wave][0];
  const unsigned long long in_base = (unsigned long long)in;
  const unsigned long long w_base  = (unsigned long long)w;

  v8f acc00 = {}, acc10 = {}, acc01 = {}, acc11 = {};

  for (int kk = 0; kk < Cin; kk += 32) {
    // weight slab (32 Cout rows x 32 K), wave 0 only
    if (wave == 0)
      tdm_load_2d(lds_w, w_base + 2ull * ((unsigned long long)mbase * Cin + kk),
                  Cin - kk, Cout - mbase, 32, 32, Cin);
    // per-wave B tile (32 K-rows x 32 positions)
    if (wvalid)
      tdm_load_2d(lds_b,
                  in_base + 2ull * ((unsigned long long)(img * Cin + kk) *
                                        (unsigned)HW + (unsigned)x0),
                  HW - x0, Cin - kk, 32, 32, HW);
    __builtin_amdgcn_s_wait_tensorcnt(0);
    __syncthreads();

    if (wvalid) {
      union { v16bf v; unsigned short u[16]; } A0, A1, B0, B1;
#pragma unroll
      for (int e = 0; e < 16; ++e) {
        const int kq = e + 8 * half + (e >= 8 ? 8 : 0);
        A0.u[e] = sW[lm * 32 + kq];
        A1.u[e] = sW[(16 + lm) * 32 + kq];
        B0.u[e] = sB[wave][kq * 32 + lm];
        B1.u[e] = sB[wave][kq * 32 + 16 + lm];
      }
      acc00 = __builtin_amdgcn_wmma_f32_16x16x32_bf16(
          false, A0.v, false, B0.v, (short)0, acc00, false, false);
      acc10 = __builtin_amdgcn_wmma_f32_16x16x32_bf16(
          false, A1.v, false, B0.v, (short)0, acc10, false, false);
      acc01 = __builtin_amdgcn_wmma_f32_16x16x32_bf16(
          false, A0.v, false, B1.v, (short)0, acc01, false, false);
      acc11 = __builtin_amdgcn_wmma_f32_16x16x32_bf16(
          false, A1.v, false, B1.v, (short)0, acc11, false, false);
    }
    __syncthreads();
  }

  if (wvalid) {
    const int p0 = x0 + lm;        // position subtile 0
    const int p1 = x0 + 16 + lm;   // position subtile 1
    const bool pv0 = p0 < HW;
    const bool pv1 = p1 < HW;
#pragma unroll
    for (int r = 0; r < 8; ++r) {
      const int m = r + 8 * half;
      const int co0 = mbase + m;
      const int co1 = mbase + 16 + m;
      if (co0 < Cout) {
        const float b = bias[co0];
        if (pv0) {
          float v = acc00[r] + b;
          if (relu) v = fmaxf(v, 0.0f);
          out[(img * Cout + co0) * HW + p0] = f32_to_bf16(v);
        }
        if (pv1) {
          float v = acc01[r] + b;
          if (relu) v = fmaxf(v, 0.0f);
          out[(img * Cout + co0) * HW + p1] = f32_to_bf16(v);
        }
      }
      if (co1 < Cout) {
        const float b = bias[co1];
        if (pv0) {
          float v = acc10[r] + b;
          if (relu) v = fmaxf(v, 0.0f);
          out[(img * Cout + co1) * HW + p0] = f32_to_bf16(v);
        }
        if (pv1) {
          float v = acc11[r] + b;
          if (relu) v = fmaxf(v, 0.0f);
          out[(img * Cout + co1) * HW + p1] = f32_to_bf16(v);
        }
      }
    }
  }
}

// ---------------------------------------------------------------------------
// Training-mode BatchNorm: stats over (N,H,W), one block per channel.
// ---------------------------------------------------------------------------
__global__ void k_bn_stats(const unsigned short* __restrict__ x,
                           const float* __restrict__ gamma,
                           const float* __restrict__ beta,
                           float* __restrict__ scale,
                           float* __restrict__ shift,
                           int Nb, int C, int HW) {
  const int c = blockIdx.x;
  const int total = Nb * HW;
  float s = 0.f, ss = 0.f;
  for (int t = threadIdx.x; t < total; t += blockDim.x) {
    const int n = t / HW;
    const int i = t - n * HW;
    const float v = bf16_to_f32(x[(n * C + c) * HW + i]);
    s += v; ss += v * v;
  }
  __shared__ float sh1[256], sh2[256];
  sh1[threadIdx.x] = s; sh2[threadIdx.x] = ss;
  __syncthreads();
  for (int wdt = 128; wdt > 0; wdt >>= 1) {
    if ((int)threadIdx.x < wdt) {
      sh1[threadIdx.x] += sh1[threadIdx.x + wdt];
      sh2[threadIdx.x] += sh2[threadIdx.x + wdt];
    }
    __syncthreads();
  }
  if (threadIdx.x == 0) {
    const float inv = 1.0f / (float)total;
    const float mean = sh1[0] * inv;
    const float var  = sh2[0] * inv - mean * mean;   // biased variance
    const float sc = gamma[c] * rsqrtf(var + 1e-5f);
    scale[c] = sc;
    shift[c] = beta[c] - mean * sc;
  }
}

__global__ void k_bn_apply_relu(const unsigned short* __restrict__ x,
                                unsigned short* __restrict__ y,
                                const float* __restrict__ scale,
                                const float* __restrict__ shift,
                                int C, int HW, int total) {
  for (int i = blockIdx.x * blockDim.x + threadIdx.x; i < total;
       i += gridDim.x * blockDim.x) {
    const int c = (i / HW) % C;
    const float v = bf16_to_f32(x[i]) * scale[c] + shift[c];
    y[i] = f32_to_bf16(fmaxf(v, 0.0f));
  }
}

// 2x2 maxpool, stride 2, VALID
__global__ void k_maxpool2(const unsigned short* __restrict__ x,
                           unsigned short* __restrict__ y,
                           int NC, int H, int W, int Ho, int Wo, int total) {
  for (int i = blockIdx.x * blockDim.x + threadIdx.x; i < total;
       i += gridDim.x * blockDim.x) {
    const int hwo = Ho * Wo;
    const int nc = i / hwo;
    const int r  = i - nc * hwo;
    const int ohh = r / Wo;
    const int oww = r - ohh * Wo;
    const unsigned short* p = x + (nc * H + ohh * 2) * W + oww * 2;
    float m = fmaxf(fmaxf(bf16_to_f32(p[0]), bf16_to_f32(p[1])),
                    fmaxf(bf16_to_f32(p[W]), bf16_to_f32(p[W + 1])));
    y[i] = f32_to_bf16(m);
  }
}

// global average pool (H,W) -> fp32 logits
__global__ void k_final_mean(const unsigned short* __restrict__ x,
                             float* __restrict__ out, int NC, int HW) {
  const int i = blockIdx.x * blockDim.x + threadIdx.x;
  if (i < NC) {
    float s = 0.f;
    for (int j = 0; j < HW; ++j) s += bf16_to_f32(x[i * HW + j]);
    out[i] = s / (float)HW;
  }
}

// ---------------------------------------------------------------------------
// Host orchestration
// ---------------------------------------------------------------------------
extern "C" void kernel_launch(void* const* d_in, const int* in_sizes, int n_in,
                              void* d_out, int out_size, void* d_ws, size_t ws_size,
                              hipStream_t stream) {
  (void)in_sizes; (void)out_size; (void)ws_size;

  const float* x = (const float*)d_in[0];
  // params flattened in dict order: block1..4, each {w1,b1,w2,b2,g2,be2,w3,b3,g3,be3}
  const float* P[40] = {nullptr};
  for (int i = 0; i < 40 && i + 1 < n_in; ++i) P[i] = (const float*)d_in[i + 1];
  auto prm = [&](int b, int f) { return P[b * 10 + f]; };

  char* base = (char*)d_ws;
  size_t off = 0;
  auto alloc = [&](size_t bytes) -> void* {
    void* p = base + off;
    off = (off + bytes + 255) & ~(size_t)255;
    return p;
  };

  unsigned short* XB = (unsigned short*)alloc((size_t)64 * 3 * 384 * 384 * 2);
  unsigned short* A  = (unsigned short*)alloc((size_t)64000000 * 2);
  unsigned short* B  = (unsigned short*)alloc((size_t)64000000 * 2);
  float* scale = (float*)alloc(512 * sizeof(float));
  float* shift = (float*)alloc(512 * sizeof(float));

  static const long wsz[12] = {
      64L * 3 * 11 * 11,  64L * 64,   64L * 64,      // block1 w1,w2,w3
      256L * 64 * 5 * 5,  256L * 256, 256L * 256,    // block2
      384L * 256 * 3 * 3, 384L * 384, 384L * 384,    // block3
      100L * 384 * 3 * 3, 100L * 100, 100L * 100};   // block4
  unsigned short* W[12];
  for (int i = 0; i < 12; ++i) W[i] = (unsigned short*)alloc((size_t)wsz[i] * 2);

  auto cvt = [&](const float* s, unsigned short* d, long n) {
    unsigned blk = (unsigned)((n + 255) / 256);
    k_cast_bf16<<<dim3(blk), dim3(256), 0, stream>>>(s, d, n);
  };

  cvt(x, XB, (long)64 * 3 * 384 * 384);
  const int wf[3] = {0, 2, 6};
  for (int b = 0; b < 4; ++b)
    for (int j = 0; j < 3; ++j) cvt(prm(b, wf[j]), W[b * 3 + j], wsz[b * 3 + j]);

  auto conv = [&](const unsigned short* in, const unsigned short* wb,
                  const float* bias, unsigned short* out,
                  int Nb, int Ci, int Hi, int Wi, int Co, int Ho, int Wo,
                  int k, int s, int p, int relu) {
    long Ntot = (long)Nb * Ho * Wo;
    dim3 g((unsigned)((Ntot + 127) / 128), (unsigned)((Co + 15) / 16));
    k_conv_wmma<<<g, dim3(256), 0, stream>>>(in, wb, bias, out,
                                             Nb, Ci, Hi, Wi, Co, Ho, Wo,
                                             k, s, p, relu);
  };
  auto gemm1x1 = [&](const unsigned short* in, const unsigned short* wb,
                     const float* bias, unsigned short* out,
                     int Nb, int Ci, int HW, int Co, int relu) {
    int tiles = Nb * ((HW + 31) / 32);
    dim3 g((unsigned)((tiles + 7) / 8), (unsigned)((Co + 31) / 32));
    k_gemm1x1_wmma<<<g, dim3(256), 0, stream>>>(in, wb, bias, out,
                                                Nb, Ci, HW, Co, relu);
  };
  auto bn = [&](const unsigned short* t, unsigned short* o,
                const float* g_, const float* b_, int Nb, int C, int HW) {
    k_bn_stats<<<dim3((unsigned)C), dim3(256), 0, stream>>>(t, g_, b_, scale,
                                                            shift, Nb, C, HW);
    int total = Nb * C * HW;
    k_bn_apply_relu<<<dim3((unsigned)((total + 255) / 256)), dim3(256), 0,
                      stream>>>(t, o, scale, shift, C, HW, total);
  };
  auto pool = [&](const unsigned short* t, unsigned short* o, int NC, int H,
                  int Wd) {
    int Ho = H / 2, Wo = Wd / 2;
    int total = NC * Ho * Wo;
    k_maxpool2<<<dim3((unsigned)((total + 255) / 256)), dim3(256), 0, stream>>>(
        t, o, NC, H, Wd, Ho, Wo, total);
  };

  // ---- block 1: 3 -> 64, 11x11 s3 p0, 384 -> 125 ----
  conv(XB, W[0], prm(0, 1), A, 64, 3, 384, 384, 64, 125, 125, 11, 3, 0, 1);
  gemm1x1(A, W[1], prm(0, 3), B, 64, 64, 125 * 125, 64, 0);
  bn(B, A, prm(0, 4), prm(0, 5), 64, 64, 125 * 125);
  gemm1x1(A, W[2], prm(0, 7), B, 64, 64, 125 * 125, 64, 0);
  bn(B, A, prm(0, 8), prm(0, 9), 64, 64, 125 * 125);
  pool(A, B, 64 * 64, 125, 125);  // -> (64,64,62,62)

  // ---- block 2: 64 -> 256, 5x5 s2 p2, 62 -> 31 ----
  conv(B, W[3], prm(1, 1), A, 64, 64, 62, 62, 256, 31, 31, 5, 2, 2, 1);
  gemm1x1(A, W[4], prm(1, 3), B, 64, 256, 31 * 31, 256, 0);
  bn(B, A, prm(1, 4), prm(1, 5), 64, 256, 31 * 31);
  gemm1x1(A, W[5], prm(1, 7), B, 64, 256, 31 * 31, 256, 0);
  bn(B, A, prm(1, 8), prm(1, 9), 64, 256, 31 * 31);
  pool(A, B, 64 * 256, 31, 31);  // -> (64,256,15,15)

  // ---- block 3: 256 -> 384, 3x3 s1 p1, 15 -> 15 ----
  conv(B, W[6], prm(2, 1), A, 64, 256, 15, 15, 384, 15, 15, 3, 1, 1, 1);
  gemm1x1(A, W[7], prm(2, 3), B, 64, 384, 225, 384, 0);
  bn(B, A, prm(2, 4), prm(2, 5), 64, 384, 225);
  gemm1x1(A, W[8], prm(2, 7), B, 64, 384, 225, 384, 0);
  bn(B, A, prm(2, 8), prm(2, 9), 64, 384, 225);
  pool(A, B, 64 * 384, 15, 15);  // -> (64,384,7,7)

  // ---- block 4: 384 -> 100, 3x3 s1 p1, 7 -> 7 ----
  conv(B, W[9], prm(3, 1), A, 64, 384, 7, 7, 100, 7, 7, 3, 1, 1, 1);
  gemm1x1(A, W[10], prm(3, 3), B, 64, 100, 49, 100, 0);
  bn(B, A, prm(3, 4), prm(3, 5), 64, 100, 49);
  gemm1x1(A, W[11], prm(3, 7), B, 64, 100, 49, 100, 0);
  bn(B, A, prm(3, 8), prm(3, 9), 64, 100, 49);

  // ---- global average pool -> fp32 logits (64,100) ----
  k_final_mean<<<dim3(25), dim3(256), 0, stream>>>(A, (float*)d_out, 6400, 49);
}
